// RNN_9414568313248
// MI455X (gfx1250) — matrix operand order repaired and checked by reading, not compiled, not run on hardware
//
#include <hip/hip_runtime.h>
#include <hip/hip_bf16.h>

typedef __attribute__((ext_vector_type(16))) _Float16 v16h;
typedef __attribute__((ext_vector_type(8)))  _Float16 v8h;
typedef __attribute__((ext_vector_type(8)))  float    v8f;
typedef __attribute__((ext_vector_type(4)))  float    v4f;

#define NS   16          // samples
#define NC   64          // channels (input size)
#define NT   128         // lookback (time steps)
#define NA   64          // assets
#define NB   (NS * NA)   // fused batch = 1024
#define HD   256         // per-direction hidden
#define NG   (4 * HD)    // gate width = 1024
#define NH   512         // bidirectional hidden

#define HSTRIDE 264      // 256 + 8 halves pad (keeps 16B align, avoids bank conflicts)
#define GSTRIDE 1032     // 1024 + 8 floats pad (kills 16-way conflict on gate reads)

#define NKT  10          // K tiles per step: 2 (input proj, K=64) + 8 (recurrent, K=256)

#define SHUF16(lo, hi) __builtin_shufflevector((lo), (hi), 0,1,2,3,4,5,6,7,8,9,10,11,12,13,14,15)

__device__ __forceinline__ float fsig(float x)  { return 1.0f / (1.0f + __expf(-x)); }
__device__ __forceinline__ float ftanh(float x) { float e = __expf(2.0f * x); return 1.0f - 2.0f / (e + 1.0f); }

// A fragment (16x32 f16): lanes 0-15 = rows, K split per ISA 16-bit A layout.
__device__ __forceinline__ v16h load_afrag(const _Float16* row, int kb, int half) {
    v8h lo = *(const v8h*)(row + kb + 8 * half);
    v8h hi = *(const v8h*)(row + kb + 16 + 8 * half);
    return SHUF16(lo, hi);
}

// 4 B fragments (32x16 f16 each): lane column n = lane%16, 16 K per lane in ISA table order.
__device__ __forceinline__ void load_b4(const _Float16* base, int kb, int lane, int nb, v16h bf[4]) {
    const int h = lane >> 4;
    const _Float16* bp = base + (size_t)kb * NG + nb + (lane & 15);
    #pragma unroll
    for (int q = 0; q < 4; ++q)
        #pragma unroll
        for (int s = 0; s < 16; ++s) {
            const int k = (s < 8) ? (8 * h + s) : (8 + 8 * h + s);
            bf[q][s] = bp[(size_t)k * NG + 16 * q];
        }
}

// ---------------------------------------------------------------------------
// x: (S, C, T, A) f32  ->  xT: (b = s*A + a, t, c) f16, LDS-tiled 64x64 (c,a)
// ---------------------------------------------------------------------------
__global__ __launch_bounds__(256) void k_transpose_x(const float* __restrict__ x,
                                                     _Float16* __restrict__ xT) {
    __shared__ _Float16 tile[64][65];
    const int blk = blockIdx.x;            // 0..2047 = S*T
    const int s = blk >> 7, t = blk & 127;
    const int tid = threadIdx.x;
    #pragma unroll
    for (int i = 0; i < 16; ++i) {
        int idx = tid + i * 256;           // 64*64 elems
        int a = idx & 63, c = idx >> 6;    // read contiguous in a
        tile[a][c] = (_Float16)x[(((size_t)(s * 64 + c)) * NT + t) * NA + a];
    }
    __syncthreads();
    #pragma unroll
    for (int i = 0; i < 16; ++i) {
        int idx = tid + i * 256;
        int c = idx & 63, a = idx >> 6;    // write contiguous in c
        xT[(((size_t)(s * 64 + a)) * NT + t) * NC + c] = tile[a][c];
    }
}

// ---------------------------------------------------------------------------
// W: (NG rows, cols) f32  ->  Wt: (cols, NG) f16   (B-operand: K-major rows)
// ---------------------------------------------------------------------------
__global__ void k_transpose_w(const float* __restrict__ W, _Float16* __restrict__ Wt, int cols) {
    int tid = blockIdx.x * blockDim.x + threadIdx.x;
    if (tid >= NG * cols) return;
    int n = tid / cols, c = tid % cols;
    Wt[(size_t)c * NG + n] = (_Float16)W[tid];
}

// ---------------------------------------------------------------------------
// Persistent-per-tile bidirectional LSTM.
// grid = (64 batch-tiles, 2 directions), block = 256 (8 wave32).
// Wave w owns gate columns [w*128, w*128+128), processed as 2 N-groups of 64
// (4 WMMA n-tiles each) to bound accumulator pressure.
// ---------------------------------------------------------------------------
__global__ __launch_bounds__(256)
void k_bilstm(const _Float16* __restrict__ xT,
              const _Float16* __restrict__ WihT_f, const _Float16* __restrict__ WhhT_f,
              const _Float16* __restrict__ WihT_b, const _Float16* __restrict__ WhhT_b,
              const float* __restrict__ bih_f, const float* __restrict__ bhh_f,
              const float* __restrict__ bih_b, const float* __restrict__ bhh_b,
              float* __restrict__ out) {
    __shared__ __align__(16) _Float16 h_sm[16 * HSTRIDE];   // 8.25 KB, A staging
    __shared__ __align__(16) float    g_sm[16 * GSTRIDE];   // 64.5 KB, gate exchange

    const int tile = blockIdx.x;        // batch tile: rows tile*16 .. +15
    const int dir  = blockIdx.y;        // 0 = forward, 1 = backward
    const int tid  = threadIdx.x;
    const int lane = tid & 31;
    const int wave = tid >> 5;          // 0..7
    const int half = lane >> 4;         // hi/lo lane group (K split per ISA layout)
    const int l16  = lane & 15;
    const int nbase = wave * 128;

    const float* bih = dir ? bih_b : bih_f;
    const float* bhh = dir ? bhh_b : bhh_f;

    // Opaque zero stride: advancing the weight pointers by zstep each step makes
    // them loop-carried PHIs (LICM cannot hoist the time-invariant B fragments
    // -> no spills) while keeping the SSA chain rooted in the kernarg pointer
    // (address-space inference keeps global_load, not flat_load).
    size_t zstep = 0;
#if defined(__HIP_DEVICE_COMPILE__)
    asm volatile("" : "+s"(zstep));
#endif
    const _Float16* WihT = dir ? WihT_b : WihT_f;
    const _Float16* WhhT = dir ? WhhT_b : WhhT_f;

    // bias per accumulator n-tile (constant across D rows and time)
    float bsum[8];
    #pragma unroll
    for (int nt = 0; nt < 8; ++nt) {
        int col = nbase + nt * 16 + l16;
        bsum[nt] = bih[col] + bhh[col];
    }

    for (int i = tid; i < 16 * HSTRIDE; i += 256) h_sm[i] = (_Float16)0.0f;

    // elementwise ownership: thread -> (row er, 16 consecutive h-cols at jb)
    const int er = tid & 15;
    const int jb = (tid >> 4) << 4;
    float cst[16];
    #pragma unroll
    for (int jj = 0; jj < 16; ++jj) cst[jj] = 0.0f;

    const int brow = tile * 16 + l16;                        // A-operand row (GEMM)
    const _Float16* xbase = xT + (size_t)brow * NT * NC;
    const _Float16* hrow  = h_sm + l16 * HSTRIDE;

    const int eb = tile * 16 + er;                           // batch row (elementwise)
    const int es = eb >> 6, ea = eb & 63;
    float* outb = out + ((size_t)es * NH + dir * HD) * NT * NA + ea;

    for (int it = 0; it < NT; ++it) {
        const int t = dir ? (NT - 1 - it) : it;
        const _Float16* xrow = xbase + (size_t)t * NC;

        __syncthreads();   // h_sm from previous step visible; prev gate reads done

        #pragma unroll 1
        for (int ng = 0; ng < 2; ++ng) {
            const int nb = nbase + ng * 64;

            union { v8f v; float f[8]; } acc[4];
            #pragma unroll
            for (int nt = 0; nt < 4; ++nt)
                #pragma unroll
                for (int r = 0; r < 8; ++r) acc[nt].f[r] = bsum[ng * 4 + nt];

            // ---- K pipeline: kt 0..1 = input proj (x), kt 2..9 = recurrent (h) ----
            v16h bf[4], bn[4];
            v16h afrag = load_afrag(xrow, 0, half);
            load_b4(WihT, 0, lane, nb, bf);

            #pragma unroll
            for (int kt = 0; kt < NKT; ++kt) {
                if (kt + 1 < NKT) {                       // prefetch next k-tile
                    const int kn = kt + 1;
                    if (kn < 2) load_b4(WihT, kn * 32, lane, nb, bn);
                    else        load_b4(WhhT, (kn - 2) * 32, lane, nb, bn);
                }
                #pragma unroll
                for (int nt = 0; nt < 4; ++nt)
                    acc[nt].v = __builtin_amdgcn_wmma_f32_16x16x32_f16(
                        false, afrag, false, bf[nt], (short)0, acc[nt].v, false, false);
                if (kt + 1 < NKT) {
                    const int kn = kt + 1;
                    afrag = (kn < 2) ? load_afrag(xrow, kn * 32, half)
                                     : load_afrag(hrow, (kn - 2) * 32, half);
                    #pragma unroll
                    for (int q = 0; q < 4; ++q) bf[q] = bn[q];
                }
            }

            // scatter gates (D layout: VGPR r -> M = r + 8*half, col = nb+nt*16+l16)
            #pragma unroll
            for (int nt = 0; nt < 4; ++nt) {
                const int col = nb + nt * 16 + l16;
                #pragma unroll
                for (int r = 0; r < 8; ++r)
                    g_sm[(r + 8 * half) * GSTRIDE + col] = acc[nt].f[r];
            }
        }

        __syncthreads();

        // ---- LSTM cell (gate order i,f,g,o); vectorized LDS access ----
        const float* gr = g_sm + er * GSTRIDE;
        v4f gi4[4], gf4[4], gg4[4], go4[4];
        #pragma unroll
        for (int q = 0; q < 4; ++q) {
            gi4[q] = *(const v4f*)(gr + jb + 4 * q);
            gf4[q] = *(const v4f*)(gr + 256 + jb + 4 * q);
            gg4[q] = *(const v4f*)(gr + 512 + jb + 4 * q);
            go4[q] = *(const v4f*)(gr + 768 + jb + 4 * q);
        }
        v8h h0, h1;
        float* orow = outb + (size_t)t * NA;
        #pragma unroll
        for (int jj = 0; jj < 16; ++jj) {
            float gi = fsig (gi4[jj >> 2][jj & 3]);
            float gf = fsig (gf4[jj >> 2][jj & 3]);
            float gg = ftanh(gg4[jj >> 2][jj & 3]);
            float go = fsig (go4[jj >> 2][jj & 3]);
            float cv = gf * cst[jj] + gi * gg;
            cst[jj] = cv;
            float hv = go * ftanh(cv);
            if (jj < 8) h0[jj] = (_Float16)hv; else h1[jj - 8] = (_Float16)hv;
            orow[(size_t)(jb + jj) * NT * NA] = hv;   // adjacent lanes -> adjacent a
        }
        *(v8h*)(h_sm + er * HSTRIDE + jb)     = h0;   // f16 h for next step's A
        *(v8h*)(h_sm + er * HSTRIDE + jb + 8) = h1;

        if (it + 1 < NT) {                            // global_prefetch_b8 next x row
            const int tn = dir ? (NT - 2 - it) : (it + 1);
            __builtin_prefetch(xbase + (size_t)tn * NC, 0, 0);
        }

        WihT += zstep;   // opaque zero: defeats LICM, keeps global address space
        WhhT += zstep;
    }
}

// ---------------------------------------------------------------------------
extern "C" void kernel_launch(void* const* d_in, const int* in_sizes, int n_in,
                              void* d_out, int out_size, void* d_ws, size_t ws_size,
                              hipStream_t stream) {
    const float* x     = (const float*)d_in[0];
    const float* Wih_f = (const float*)d_in[1];
    const float* Whh_f = (const float*)d_in[2];
    const float* bih_f = (const float*)d_in[3];
    const float* bhh_f = (const float*)d_in[4];
    const float* Wih_b = (const float*)d_in[5];
    const float* Whh_b = (const float*)d_in[6];
    const float* bih_b = (const float*)d_in[7];
    const float* bhh_b = (const float*)d_in[8];

    // workspace (f16): xT 16 MB | WihT_f 128K | WihT_b 128K | WhhT_f 512K | WhhT_b 512K
    char* ws = (char*)d_ws;
    size_t off = 0;
    _Float16* xT     = (_Float16*)(ws + off); off += (size_t)NB * NT * NC * 2;
    _Float16* WihT_f = (_Float16*)(ws + off); off += (size_t)NC * NG * 2;
    _Float16* WihT_b = (_Float16*)(ws + off); off += (size_t)NC * NG * 2;
    _Float16* WhhT_f = (_Float16*)(ws + off); off += (size_t)HD * NG * 2;
    _Float16* WhhT_b = (_Float16*)(ws + off); off += (size_t)HD * NG * 2;
    (void)ws_size; (void)in_sizes; (void)n_in; (void)out_size;

    k_transpose_x<<<NS * NT, 256, 0, stream>>>(x, xT);
    k_transpose_w<<<(NG * NC + 255) / 256, 256, 0, stream>>>(Wih_f, WihT_f, NC);
    k_transpose_w<<<(NG * NC + 255) / 256, 256, 0, stream>>>(Wih_b, WihT_b, NC);
    k_transpose_w<<<(NG * HD + 255) / 256, 256, 0, stream>>>(Whh_f, WhhT_f, HD);
    k_transpose_w<<<(NG * HD + 255) / 256, 256, 0, stream>>>(Whh_b, WhhT_b, HD);

    dim3 grid(NB / 16, 2);
    k_bilstm<<<grid, 256, 0, stream>>>(xT, WihT_f, WhhT_f, WihT_b, WhhT_b,
                                       bih_f, bhh_f, bih_b, bhh_b, (float*)d_out);
}
